// DNMF_45449343926614
// MI455X (gfx1250) — matrix operand order repaired and checked by reading, not compiled, not run on hardware
//
#include <hip/hip_runtime.h>
#include <hip/hip_bf16.h>
#include <cstdint>
#include <cstddef>

// ---------------------------------------------------------------------------
// DNMF community detection on gfx1250 (MI455X).
// N=2048, K=16. All big GEMMs are NT-form bf16 WMMA (exact for binary A;
// Newton-Schulz inverse in bf16 matrix / f32 accum; 16x16 polar in f32). Working set
// (~64MB) fits in the 192MB L2 -> compute-bound on the WMMA pipes.
// B-tiles staged into LDS by the Tensor Data Mover (TDM), double buffered.
// ---------------------------------------------------------------------------

#define N_DIM     2048
#define KC        16
#define ALPHA_C   0.5f
#define BETA_C    0.1f
#define GAMMA_C   0.1f
#define EPS_DEN   1e-15f
#define NS_ITERS  30
#define POLAR_IT  24
#define NUM_OUTER 2
#define NUM_INNER 2

typedef __bf16 bf16;
typedef bf16  v16bf __attribute__((ext_vector_type(16)));
typedef bf16  v8bf  __attribute__((ext_vector_type(8)));
typedef float v8f   __attribute__((ext_vector_type(8)));
typedef float v4f   __attribute__((ext_vector_type(4)));

typedef uint32_t u32x4 __attribute__((ext_vector_type(4)));
typedef int      i32x8 __attribute__((ext_vector_type(8)));
typedef int      i32x4 __attribute__((ext_vector_type(4)));

// Exact pointee type for the async global->LDS builtins:
//   void (V4i addrspace(1)*, V4i addrspace(3)*, imm offset, imm cpol)
typedef int v4i32 __attribute__((vector_size(16)));
typedef __attribute__((address_space(1))) v4i32 gv4i32;
typedef __attribute__((address_space(3))) v4i32 lv4i32;
typedef __attribute__((address_space(3))) void  las_void;

// Staging mode: 2 = TDM tensor_load_to_lds, 1 = async global->LDS, 0 = sync.
#if __has_builtin(__builtin_amdgcn_tensor_load_to_lds) && \
    __has_builtin(__builtin_amdgcn_s_wait_tensorcnt)
#define GEMM_STAGE 2
#pragma message("ATHENA: GEMM staging via TDM tensor_load_to_lds")
#elif __has_builtin(__builtin_amdgcn_global_load_async_to_lds_b128) && \
      __has_builtin(__builtin_amdgcn_s_wait_asynccnt)
#define GEMM_STAGE 1
#pragma message("ATHENA: GEMM staging via GLOBAL_LOAD_ASYNC_TO_LDS_B128")
#else
#define GEMM_STAGE 0
#pragma message("ATHENA: GEMM staging via sync load + ds_store fallback")
#endif

// This toolchain (clang >= 23) exposes the 6-arg tensor_load_to_lds.
#if defined(__clang_major__) && __clang_major__ >= 23
#define TDM_LOAD(g0, g1, g2, g3, gz8) \
  __builtin_amdgcn_tensor_load_to_lds(g0, g1, g2, g3, gz8, 0)
#else
#define TDM_LOAD(g0, g1, g2, g3, gz8) \
  __builtin_amdgcn_tensor_load_to_lds(g0, g1, g2, g3, 0)
#endif

__device__ __forceinline__ v16bf pack16(v8bf lo, v8bf hi) {
  v16bf r;
  #pragma unroll
  for (int i = 0; i < 8; ++i) { r[i] = lo[i]; r[i + 8] = hi[i]; }
  return r;
}
// A fragment: lane holds row (lane&15); K chunks {0..7,16..23} (lanes<16) or
// {8..15,24..31} (lanes>=16) -> two 128-bit loads, second at +16 elements.
__device__ __forceinline__ v16bf load_afrag(const bf16* p) {
  return pack16(*(const v8bf*)p, *(const v8bf*)(p + 16));
}
// B fragment (NT): lane holds row (lane&15); same K chunks as A fragment.
__device__ __forceinline__ v16bf load_bfrag(const bf16* p) {
  return pack16(*(const v8bf*)p, *(const v8bf*)(p + 16));
}

// ---------------------------------------------------------------------------
// NT GEMM: C[M x Nc] = A[M x Kd] @ B[Nc x Kd]^T (bf16 in, f32 out; optional
// fused bf16 copy). Block = 256 threads = 8 waves. Wave owns a
// (16*RPW) x (16*CPW) tile: RPW*CPW accumulators, each A fragment feeds CPW
// WMMAs. B-tile (16*CPW rows x 256 K, padded rows) double-buffered in LDS,
// staged by one TDM descriptor per chunk (wave 0), one barrier per 8 K-steps.
// ---------------------------------------------------------------------------
template <int RPW, int CPW>
__global__ __launch_bounds__(256)
void k_gemm_nt_bf16(const bf16* __restrict__ A, const bf16* __restrict__ B,
                    float* __restrict__ C, bf16* __restrict__ Cb,
                    int M, int Nc, int Kd) {
  constexpr int SROWS = 16 * CPW;     // B rows staged per chunk
  constexpr int KCH   = 256;          // staged K elements per chunk
  constexpr int LROW  = 264;          // LDS row stride in bf16 (528B, padded)
  __shared__ bf16 sB[2][SROWS * LROW];

  const int t    = threadIdx.x;
  const int wave = t >> 5;
  const int lane = t & 31;
  const int m0   = (blockIdx.y * 8 + wave) * (16 * RPW);
  const int c0   = blockIdx.x * (16 * CPW);
  const int lrow = lane & 15;
  const int hsel = lane >> 4;

  const bf16* aptr = A + (size_t)(m0 + lrow) * Kd + hsel * 8;

#if GEMM_STAGE == 2
  const unsigned lds_base = (unsigned)(uintptr_t)(las_void*)&sB[0][0];
  // D# group1: data_size=2B; pad_enable, pad_interval=128 DW, pad_amount=4 DW
  // (row stride 512B+16B = 528B); tensor = Nc x Kd; tile = SROWS x 256.
  const i32x8 g1v = {
    (int)((1u << 16) | (1u << 20) | (6u << 22) | (3u << 25)),
    (int)(((uint32_t)Kd & 0xFFFFu) << 16),
    (int)((((uint32_t)Kd >> 16) & 0xFFFFu) | (((uint32_t)Nc & 0xFFFFu) << 16)),
    (int)((((uint32_t)Nc >> 16) & 0xFFFFu) | ((uint32_t)KCH << 16)),
    (int)(uint32_t)SROWS,
    (int)(uint32_t)Kd,
    0, 0 };
  const i32x4 gz4 = { 0, 0, 0, 0 };
  const i32x8 gz8 = { 0, 0, 0, 0, 0, 0, 0, 0 };
#endif

  auto stage = [&](int buf, int k0) {
#if GEMM_STAGE == 2
    if (wave == 0) {
      uint64_t ga = (uint64_t)(uintptr_t)(B + (size_t)c0 * Kd + k0);
      u32x4 g0 = { 1u,
                   lds_base + (unsigned)buf * (unsigned)(SROWS * LROW * 2),
                   (uint32_t)ga,
                   (uint32_t)((ga >> 32) & 0x01FFFFFFu) | (2u << 30) };
      TDM_LOAD(g0, g1v, gz4, gz4, gz8);
    }
#else
    for (int c = t; c < SROWS * 32; c += 256) {
      const int r = c >> 5, q = c & 31;
      const bf16* src = B + (size_t)(c0 + r) * Kd + k0 + q * 8;
      bf16* dst = &sB[buf][r * LROW + q * 8];
#if GEMM_STAGE == 1
      __builtin_amdgcn_global_load_async_to_lds_b128(
          (gv4i32*)src, (lv4i32*)dst, 0, 0);
#else
      *(v8bf*)dst = *(const v8bf*)src;
#endif
    }
#endif
  };
  auto wait_stage = [&]() {
#if GEMM_STAGE == 2
    if (wave == 0) __builtin_amdgcn_s_wait_tensorcnt(0);
#elif GEMM_STAGE == 1
    __builtin_amdgcn_s_wait_asynccnt(0);
#endif
    __syncthreads();
  };

  v8f acc[RPW][CPW];
  #pragma unroll
  for (int r = 0; r < RPW; ++r)
    #pragma unroll
    for (int c = 0; c < CPW; ++c)
      #pragma unroll
      for (int q = 0; q < 8; ++q) acc[r][c][q] = 0.f;

  stage(0, 0);
  wait_stage();
  const int nch = Kd / KCH;
  for (int ch = 0; ch < nch; ++ch) {
    const int k0  = ch * KCH;
    const int cur = ch & 1;
    if (ch + 1 < nch) stage(cur ^ 1, k0 + KCH);   // TDM/async; overlaps WMMAs

    #pragma unroll
    for (int r = 0; r < RPW; ++r)                  // speculative prefetch OK
      __builtin_prefetch(aptr + (size_t)(r * 16) * Kd + k0 + KCH, 0, 1);

    #pragma unroll
    for (int ks = 0; ks < KCH / 32; ++ks) {
      const bf16* bbase = &sB[cur][lrow * LROW + ks * 32 + hsel * 8];
      v16bf bf[CPW];
      #pragma unroll
      for (int c = 0; c < CPW; ++c) bf[c] = load_bfrag(bbase + c * 16 * LROW);
      v16bf af[RPW];
      #pragma unroll
      for (int r = 0; r < RPW; ++r)
        af[r] = load_afrag(aptr + (size_t)(r * 16) * Kd + k0 + ks * 32);
      #pragma unroll
      for (int r = 0; r < RPW; ++r)
        #pragma unroll
        for (int c = 0; c < CPW; ++c)
          acc[r][c] = __builtin_amdgcn_wmma_f32_16x16x32_bf16(
              false, af[r], false, bf[c], (short)0, acc[r][c], false, false);
    }
    wait_stage();
  }

  // C/D layout: VGPR q, lanes 0-15 -> row +q, lanes 16-31 -> row +8+q.
  #pragma unroll
  for (int r = 0; r < RPW; ++r) {
    #pragma unroll
    for (int c = 0; c < CPW; ++c) {
      const size_t rowb = (size_t)(m0 + r * 16 + hsel * 8);
      const int    colb = c0 + c * 16 + lrow;
      float* cp = C + rowb * Nc + colb;
      #pragma unroll
      for (int q = 0; q < 8; ++q) cp[(size_t)q * Nc] = acc[r][c][q];
      if (Cb) {
        bf16* cbp = Cb + rowb * Nc + colb;
        #pragma unroll
        for (int q = 0; q < 8; ++q) cbp[(size_t)q * Nc] = (bf16)acc[r][c][q];
      }
    }
  }
}

// ---------------------------------------------------------------------------
// Elementwise / reduction helpers
// ---------------------------------------------------------------------------
__global__ void k_cast_f2b(const float* __restrict__ s, bf16* __restrict__ d,
                           size_t cnt) {
  size_t i = (size_t)blockIdx.x * 256 + threadIdx.x;
  if (i < cnt) d[i] = (bf16)s[i];
}

__global__ __launch_bounds__(256)
void k_row_sumsq(const float* __restrict__ A, float* __restrict__ out, int n) {
  __shared__ float red[256];
  int i = blockIdx.x, t = threadIdx.x;
  float s = 0.f;
  for (int j = t; j < n; j += 256) { float v = A[(size_t)i * n + j]; s += v * v; }
  red[t] = s; __syncthreads();
  for (int k = 128; k > 0; k >>= 1) { if (t < k) red[t] += red[t + k]; __syncthreads(); }
  if (t == 0) out[i] = red[0];
}

__global__ void k_kmat(float* __restrict__ G, const float* __restrict__ r) {
  size_t idx = (size_t)blockIdx.x * 256 + threadIdx.x;
  int i = (int)(idx >> 11), j = (int)(idx & 2047);
  float g = G[idx];
  G[idx] = expf(-0.5f * (r[i] + r[j] - 2.f * g));
}

__global__ __launch_bounds__(256)
void k_row_mean(const float* __restrict__ G, float* __restrict__ rm, int n) {
  __shared__ float red[256];
  int i = blockIdx.x, t = threadIdx.x;
  float s = 0.f;
  for (int j = t; j < n; j += 256) s += G[(size_t)i * n + j];
  red[t] = s; __syncthreads();
  for (int k = 128; k > 0; k >>= 1) { if (t < k) red[t] += red[t + k]; __syncthreads(); }
  if (t == 0) rm[i] = red[0] / n;
}

__global__ void k_vec_mean(const float* __restrict__ v, float* __restrict__ scal,
                           int n) {
  __shared__ float red[256];
  int t = threadIdx.x;
  float s = 0.f;
  for (int j = t; j < n; j += 256) s += v[j];
  red[t] = s; __syncthreads();
  for (int k = 128; k > 0; k >>= 1) { if (t < k) red[t] += red[t + k]; __syncthreads(); }
  if (t == 0) scal[0] = red[0] / n;
}

__global__ __launch_bounds__(256)
void k_center_row(float* __restrict__ G, const float* __restrict__ rm,
                  float* __restrict__ rs, const float* __restrict__ scal, int n) {
  __shared__ float red[256];
  int i = blockIdx.x, t = threadIdx.x;
  float tm = scal[0], ri = rm[i];
  float s = 0.f;
  for (int j = t; j < n; j += 256) {
    float v = G[(size_t)i * n + j] - ri - rm[j] + tm;
    G[(size_t)i * n + j] = v;
    s += fabsf(v + (j == i ? GAMMA_C : 0.f));
  }
  red[t] = s; __syncthreads();
  for (int k = 128; k > 0; k >>= 1) { if (t < k) red[t] += red[t + k]; __syncthreads(); }
  if (t == 0) rs[i] = red[0];
}

__global__ void k_ns_scale(const float* __restrict__ rs, float* __restrict__ scal,
                           int n) {
  __shared__ float red[256];
  int t = threadIdx.x;
  float s = 0.f;
  for (int j = t; j < n; j += 256) s = fmaxf(s, rs[j]);
  red[t] = s; __syncthreads();
  for (int k = 128; k > 0; k >>= 1) { if (t < k) red[t] = fmaxf(red[t], red[t + k]); __syncthreads(); }
  if (t == 0) { scal[1] = red[0]; scal[2] = 1.f / (red[0] * red[0] + 1e-30f); }
}

__global__ void k_prep_ns(const float* __restrict__ G, bf16* __restrict__ Mbf,
                          float* __restrict__ X, bf16* __restrict__ Xbf,
                          const float* __restrict__ scal) {
  size_t idx = (size_t)blockIdx.x * 256 + threadIdx.x;
  int i = (int)(idx >> 11), j = (int)(idx & 2047);
  float m = G[idx] + (i == j ? GAMMA_C : 0.f);
  Mbf[idx] = (bf16)m;
  float x = scal[2] * m;
  X[idx] = x;
  Xbf[idx] = (bf16)x;
}

__global__ void k_ns_update(float* __restrict__ X, const float* __restrict__ W,
                            bf16* __restrict__ Xbf) {
  size_t idx = (size_t)blockIdx.x * 256 + threadIdx.x;
  float xn = 2.f * X[idx] - W[idx];
  X[idx] = xn;
  Xbf[idx] = (bf16)xn;
}

__global__ void k_zero(float* __restrict__ p, int cnt) {
  int i = blockIdx.x * 256 + threadIdx.x;
  if (i < cnt) p[i] = 0.f;
}

__global__ __launch_bounds__(256)
void k_ata16(const float* __restrict__ A, const float* __restrict__ B,
             float* __restrict__ out, int n) {
  int t = threadIdx.x, l1 = t >> 4, l2 = t & 15;
  int r0 = blockIdx.x * (n / 16);
  float s = 0.f;
  for (int r = r0; r < r0 + n / 16; ++r)
    s += A[(size_t)r * 16 + l1] * B[(size_t)r * 16 + l2];
  atomicAdd(&out[t], s);
}

__global__ void k_nk16(const float* __restrict__ A, const float* __restrict__ B16,
                       float* __restrict__ Out, float alpha, int tB, int rows) {
  int idx = blockIdx.x * 256 + threadIdx.x;
  if (idx >= rows * 16) return;
  int i = idx >> 4, l = idx & 15;
  float s = 0.f;
  #pragma unroll
  for (int m = 0; m < 16; ++m)
    s += A[(size_t)i * 16 + m] * (tB ? B16[l * 16 + m] : B16[m * 16 + l]);
  Out[idx] = alpha * s;
}

__global__ void k_qpm(const float* __restrict__ Q, float* __restrict__ Qp,
                      float* __restrict__ Qm) {
  int t = threadIdx.x;
  float q = Q[t], a = fabsf(q);
  Qp[t] = 0.5f * (a + q);
  Qm[t] = 0.5f * (a - q);
}

__global__ void k_transpose_u(const float* __restrict__ U, bf16* __restrict__ Utb,
                              int n) {
  int idx = blockIdx.x * 256 + threadIdx.x;
  int i = idx >> 4, l = idx & 15;
  Utb[(size_t)l * n + i] = (bf16)U[idx];
}

__global__ void k_upd_u(float* __restrict__ U, const float* __restrict__ AU,
                        const float* __restrict__ FQp, const float* __restrict__ FQm,
                        const float* __restrict__ UU) {
  int idx = blockIdx.x * 256 + threadIdx.x;
  float u = U[idx];
  float num = 2.f * AU[idx] + FQp[idx];
  float den = fmaxf(2.f * UU[idx] + ALPHA_C * u + FQm[idx], EPS_DEN);
  U[idx] = u * sqrtf(sqrtf(num / den));
}

// Newton-Schulz polar factor of a 16x16 matrix (replaces U@Vh SVD products).
__global__ __launch_bounds__(256)
void k_polar16(const float* __restrict__ Min, float* __restrict__ Qout,
               int transpose_out) {
  __shared__ float Y[256], Tm[256], red[256];
  int t = threadIdx.x, i = t >> 4, j = t & 15;
  float y = Min[t];
  red[t] = y * y; __syncthreads();
  for (int s = 128; s > 0; s >>= 1) { if (t < s) red[t] += red[t + s]; __syncthreads(); }
  Y[t] = y * rsqrtf(red[0] + 1e-30f);
  __syncthreads();
  for (int it = 0; it < POLAR_IT; ++it) {
    float s = 0.f;
    #pragma unroll
    for (int m = 0; m < 16; ++m) s += Y[m * 16 + i] * Y[m * 16 + j];
    Tm[t] = s; __syncthreads();
    float s2 = 0.f;
    #pragma unroll
    for (int m = 0; m < 16; ++m)
      s2 += Y[i * 16 + m] * ((m == j ? 3.f : 0.f) - Tm[m * 16 + j]);
    __syncthreads();
    Y[t] = 0.5f * s2;
    __syncthreads();
  }
  Qout[t] = transpose_out ? Y[j * 16 + i] : Y[t];
}

// Gauss-Seidel sweep of update_f; S' never materialized:
//   S'[i,j] = beta*gamma*X[i,j] - beta/n + alpha*(i==j)
__global__ __launch_bounds__(1024)
void k_gs_sweep(const float* __restrict__ X, const float* __restrict__ UQ,
                float* __restrict__ F, int n) {
  const int t = threadIdx.x, lane = t & 31, wv = t >> 5;
  __shared__ float cs[16];
  __shared__ float wacc[32][16];
  if (t < 16) cs[t] = 0.f;
  __syncthreads();
  {
    float c[16];
    #pragma unroll
    for (int l = 0; l < 16; ++l) c[l] = 0.f;
    for (int j = t; j < n; j += 1024) {
      const v4f* fr = (const v4f*)(F + (size_t)j * 16);
      v4f f0 = fr[0], f1 = fr[1], f2 = fr[2], f3 = fr[3];
      c[0] += f0[0]; c[1] += f0[1]; c[2] += f0[2]; c[3] += f0[3];
      c[4] += f1[0]; c[5] += f1[1]; c[6] += f1[2]; c[7] += f1[3];
      c[8] += f2[0]; c[9] += f2[1]; c[10] += f2[2]; c[11] += f2[3];
      c[12] += f3[0]; c[13] += f3[1]; c[14] += f3[2]; c[15] += f3[3];
    }
    #pragma unroll
    for (int l = 0; l < 16; ++l) {
      c[l] += __shfl_xor(c[l], 16); c[l] += __shfl_xor(c[l], 8);
      c[l] += __shfl_xor(c[l], 4);  c[l] += __shfl_xor(c[l], 2);
      c[l] += __shfl_xor(c[l], 1);
    }
    if (lane == 0)
      for (int l = 0; l < 16; ++l) atomicAdd(&cs[l], c[l]);
  }
  __syncthreads();

  const float invn = 1.f / n;
  const float BG = BETA_C * GAMMA_C;
  for (int i = 0; i < n; ++i) {
    float a[16];
    #pragma unroll
    for (int l = 0; l < 16; ++l) a[l] = 0.f;
    const float* xrow = X + (size_t)i * n;
    for (int j = t; j < n; j += 1024) {
      float x = xrow[j];
      const v4f* fr = (const v4f*)(F + (size_t)j * 16);
      v4f f0 = fr[0], f1 = fr[1], f2 = fr[2], f3 = fr[3];
      a[0] += x * f0[0]; a[1] += x * f0[1]; a[2] += x * f0[2]; a[3] += x * f0[3];
      a[4] += x * f1[0]; a[5] += x * f1[1]; a[6] += x * f1[2]; a[7] += x * f1[3];
      a[8] += x * f2[0]; a[9] += x * f2[1]; a[10] += x * f2[2]; a[11] += x * f2[3];
      a[12] += x * f3[0]; a[13] += x * f3[1]; a[14] += x * f3[2]; a[15] += x * f3[3];
    }
    #pragma unroll
    for (int l = 0; l < 16; ++l) {
      a[l] += __shfl_xor(a[l], 16); a[l] += __shfl_xor(a[l], 8);
      a[l] += __shfl_xor(a[l], 4);  a[l] += __shfl_xor(a[l], 2);
      a[l] += __shfl_xor(a[l], 1);
    }
    if (lane == 0) {
      #pragma unroll
      for (int l = 0; l < 16; ++l) wacc[wv][l] = a[l];
    }
    __syncthreads();
    if (t < 16) {
      float s = 0.f;
      #pragma unroll
      for (int wq = 0; wq < 32; ++wq) s += wacc[wq][t];
      float di   = BG * xrow[i] - BETA_C * invn + ALPHA_C;
      float fil  = F[(size_t)i * 16 + t];
      float srow = BG * s - BETA_C * invn * cs[t] + ALPHA_C * fil;
      float e = di + 2.f * (srow - di * fil - ALPHA_C * UQ[(size_t)i * 16 + t]);
      float mn = e;
      mn = fminf(mn, __shfl_xor(mn, 8));
      mn = fminf(mn, __shfl_xor(mn, 4));
      mn = fminf(mn, __shfl_xor(mn, 2));
      mn = fminf(mn, __shfl_xor(mn, 1));
      float nr = (e == mn || e < 0.f) ? 1.f : 0.f;
      F[(size_t)i * 16 + t] = nr;
      cs[t] += nr - fil;
    }
    __threadfence_block();
    __syncthreads();
  }
}

// ---------------------------------------------------------------------------
extern "C" void kernel_launch(void* const* d_in, const int* in_sizes, int n_in,
                              void* d_out, int out_size, void* d_ws, size_t ws_size,
                              hipStream_t stream) {
  (void)in_sizes; (void)n_in; (void)out_size; (void)ws_size;
  const float* A  = (const float*)d_in[0];
  const float* U0 = (const float*)d_in[1];
  const float* F0 = (const float*)d_in[2];
  const float* Q0 = (const float*)d_in[3];
  const int n = N_DIM;
  const size_t nn = (size_t)n * n;

  char* w = (char*)d_ws;
  auto alloc = [&](size_t bytes) -> char* {
    char* p = w; w += (bytes + 255) & ~(size_t)255; return p;
  };
  bf16*  Abf = (bf16*)alloc(nn * 2);
  bf16*  Mbf = (bf16*)alloc(nn * 2);
  bf16*  Xbf = (bf16*)alloc(nn * 2);
  bf16*  Tbf = (bf16*)alloc(nn * 2);
  float* Xf  = (float*)alloc(nn * 4);
  float* Tf  = (float*)alloc(nn * 4);
  float* Gf  = (float*)alloc(nn * 4);
  float* rvec = (float*)alloc((size_t)n * 4);
  float* rm   = (float*)alloc((size_t)n * 4);
  float* rs   = (float*)alloc((size_t)n * 4);
  float* scal = (float*)alloc(16 * 4);
  float* U    = (float*)alloc((size_t)n * KC * 4);
  float* F    = (float*)alloc((size_t)n * KC * 4);
  bf16*  Utb  = (bf16*)alloc((size_t)n * KC * 2);
  float* AU   = (float*)alloc((size_t)n * KC * 4);
  float* FQp  = (float*)alloc((size_t)n * KC * 4);
  float* FQm  = (float*)alloc((size_t)n * KC * 4);
  float* UQ   = (float*)alloc((size_t)n * KC * 4);
  float* UU   = (float*)alloc((size_t)n * KC * 4);
  float* G16  = (float*)alloc(256 * 4);
  float* Q    = (float*)alloc(256 * 4);
  float* Qp   = (float*)alloc(256 * 4);
  float* Qm   = (float*)alloc(256 * 4);

  (void)hipMemcpyAsync(U, U0, (size_t)n * KC * 4, hipMemcpyDeviceToDevice, stream);
  (void)hipMemcpyAsync(F, F0, (size_t)n * KC * 4, hipMemcpyDeviceToDevice, stream);

  const int gEW = (int)(nn / 256);
  const int gNK = (n * KC) / 256;
  dim3 gemmSq(n / 32, n / 512);              // RPW=4,CPW=2: 512x32 per block
  dim3 gemmNK(1, n / 128);                   // RPW=1,CPW=1: 128x16 per block

  // --- Kernel-matrix path: G = A@A^T (bf16 WMMA, exact for binary A) ---
  k_cast_f2b<<<gEW, 256, 0, stream>>>(A, Abf, nn);
  k_row_sumsq<<<n, 256, 0, stream>>>(A, rvec, n);
  k_gemm_nt_bf16<4, 2><<<gemmSq, 256, 0, stream>>>(Abf, Abf, Gf, nullptr, n, n, n);
  k_kmat<<<gEW, 256, 0, stream>>>(Gf, rvec);
  k_row_mean<<<n, 256, 0, stream>>>(Gf, rm, n);
  k_vec_mean<<<1, 256, 0, stream>>>(rm, scal, n);
  k_center_row<<<n, 256, 0, stream>>>(Gf, rm, rs, scal, n);
  k_ns_scale<<<1, 256, 0, stream>>>(rs, scal, n);
  k_prep_ns<<<gEW, 256, 0, stream>>>(Gf, Mbf, Xf, Xbf, scal);

  // --- Newton-Schulz inverse: X <- 2X - X M X (all operands symmetric) ---
  for (int it = 0; it < NS_ITERS; ++it) {
    k_gemm_nt_bf16<4, 2><<<gemmSq, 256, 0, stream>>>(Xbf, Mbf, Tf, Tbf, n, n, n);
    k_gemm_nt_bf16<4, 2><<<gemmSq, 256, 0, stream>>>(Tbf, Xbf, Tf, nullptr, n, n, n);
    k_ns_update<<<gEW, 256, 0, stream>>>(Xf, Tf, Xbf);
  }
  // S = gamma*X - 1/n (implicit; consumed directly by k_gs_sweep)

  k_polar16<<<1, 256, 0, stream>>>(Q0, Q, 0);

  for (int outer = 0; outer < NUM_OUTER + 1; ++outer) {
    k_qpm<<<1, 256, 0, stream>>>(Q, Qp, Qm);
    k_nk16<<<gNK, 256, 0, stream>>>(F, Qp, FQp, ALPHA_C, 0, n);
    k_nk16<<<gNK, 256, 0, stream>>>(F, Qm, FQm, ALPHA_C, 0, n);
    for (int inner = 0; inner < 10 * NUM_INNER; ++inner) {
      k_transpose_u<<<gNK, 256, 0, stream>>>(U, Utb, n);
      k_gemm_nt_bf16<1, 1><<<gemmNK, 256, 0, stream>>>(Abf, Utb, AU, nullptr,
                                                       n, KC, n);           // A@U
      k_zero<<<1, 256, 0, stream>>>(G16, 256);
      k_ata16<<<16, 256, 0, stream>>>(U, U, G16, n);                        // U^T U
      k_nk16<<<gNK, 256, 0, stream>>>(U, G16, UU, 1.0f, 0, n);              // U(U^T U)
      k_upd_u<<<gNK, 256, 0, stream>>>(U, AU, FQp, FQm, UU);
    }
    k_nk16<<<gNK, 256, 0, stream>>>(U, Q, UQ, 1.0f, 1, n);                  // U Q^T
    for (int s = 0; s < NUM_INNER; ++s)
      k_gs_sweep<<<1, 1024, 0, stream>>>(Xf, UQ, F, n);
    k_zero<<<1, 256, 0, stream>>>(G16, 256);
    k_ata16<<<16, 256, 0, stream>>>(U, F, G16, n);                          // U^T F
    k_polar16<<<1, 256, 0, stream>>>(G16, Q, 1);   // Q ~ polar(U^T F)^T
  }

  (void)hipMemcpyAsync(d_out, F, (size_t)n * KC * 4, hipMemcpyDeviceToDevice, stream);
}
